// MambaBlock_16475494547741
// MI455X (gfx1250) — matrix operand order repaired and checked by reading, not compiled, not run on hardware
//
#include <hip/hip_runtime.h>
#include <hip/hip_bf16.h>
#include <math.h>

// ---------------- problem constants (match reference) ----------------
#define DMODEL 1024
#define DINNER 4096
#define DSTATE 16
#define DCONV  4
#define DTRANK 64
#define BATCH  2
#define SEQLEN 2048
#define BL     (BATCH * SEQLEN)      // 4096 tokens
#define NXZ    (2 * DINNER)          // 8192
#define NPROJ  (DTRANK + 2 * DSTATE) // 96
#define NCHUNK 16
#define CHUNK  (SEQLEN / NCHUNK)     // 128

// ---------------- vector types ----------------
typedef __attribute__((ext_vector_type(16))) __bf16          v16bf;
typedef __attribute__((ext_vector_type(16))) unsigned short  v16u;
typedef __attribute__((ext_vector_type(8)))  float           v8f;
typedef __attribute__((ext_vector_type(8)))  unsigned int    v8u;
typedef __attribute__((ext_vector_type(4)))  unsigned int    v4u;

__device__ __forceinline__ unsigned short f32_to_bf16_rne(float f) {
    unsigned int u = __float_as_uint(f);
    u += 0x7FFFu + ((u >> 16) & 1u);
    return (unsigned short)(u >> 16);
}
__device__ __forceinline__ float bf16_to_f32(unsigned short h) {
    return __uint_as_float(((unsigned int)h) << 16);
}

enum { EPI_F32 = 0, EPI_BF16 = 1, EPI_SOFTPLUS_BIAS = 2 };

// ---------------------------------------------------------------------
// A-fragment loader (ISA 7.12.2 A layout: e<8 -> K=hi*8+e ; e>=8 -> 16+hi*8+e-8)
// ---------------------------------------------------------------------
template <bool ABF16>
__device__ __forceinline__ v16bf load_afrag(const void* A, int lda, int row,
                                            int k0, int hi)
{
    v8u au;
    if (ABF16) {
        const unsigned short* Arow = (const unsigned short*)A + (size_t)row * lda;
        v4u lo  = *(const v4u*)(Arow + k0 + hi * 8);
        v4u hi4 = *(const v4u*)(Arow + k0 + 16 + hi * 8);
        au[0] = lo[0];  au[1] = lo[1];  au[2] = lo[2];  au[3] = lo[3];
        au[4] = hi4[0]; au[5] = hi4[1]; au[6] = hi4[2]; au[7] = hi4[3];
    } else {
        const float* Arow = (const float*)A + (size_t)row * lda;
        const float* p0 = Arow + k0 + hi * 8;
        const float* p1 = Arow + k0 + 16 + hi * 8;
        v16u at;
        #pragma unroll
        for (int e = 0; e < 8; ++e) at[e] = f32_to_bf16_rne(p0[e]);
        #pragma unroll
        for (int e = 0; e < 8; ++e) at[8 + e] = f32_to_bf16_rne(p1[e]);
        au = __builtin_bit_cast(v8u, at);
    }
    return __builtin_bit_cast(v16bf, au);
}

// ---------------------------------------------------------------------
// bf16 WMMA GEMM:  C[M,N] = A[M,K] @ Wt[N,K]^T
// Wave tile 32(M)x16(N): two accumulators share one B fragment.
// Block = 8 waves, 4(M) x 2(N) -> block tile 128M x 32N.
// Requires M%128==0, N%32==0, K%32==0.
// ---------------------------------------------------------------------
template <int EPI, bool ABF16>
__global__ __launch_bounds__(256) void wmma_gemm(
    const void* __restrict__ A, int lda,
    const unsigned short* __restrict__ Bt, int ldb,
    void* __restrict__ C, int ldc,
    const float* __restrict__ bias, int K)
{
    const int wave = threadIdx.x >> 5;
    const int lane = threadIdx.x & 31;
    const int lr   = lane & 15;
    const int hi   = lane >> 4;
    const int m0   = blockIdx.y * 128 + (wave >> 1) * 32;
    const int n0   = blockIdx.x * 32  + (wave & 1) * 16;

    const unsigned short* Brow = Bt + (size_t)(n0 + lr) * ldb;

    v8f acc0 = {}, acc1 = {};
    #pragma unroll 2
    for (int k0 = 0; k0 < K; k0 += 32) {
        v16bf a0 = load_afrag<ABF16>(A, lda, m0 + lr,      k0, hi);
        v16bf a1 = load_afrag<ABF16>(A, lda, m0 + 16 + lr, k0, hi);
        // B frag (lane col n=lr, elem e<8 -> K=hi*8+e ; e>=8 -> 16+hi*8+e-8): two 16B runs
        v4u blo = *(const v4u*)(Brow + k0 + hi * 8);
        v4u bhi = *(const v4u*)(Brow + k0 + 16 + hi * 8);
        v8u bu;
        bu[0] = blo[0]; bu[1] = blo[1]; bu[2] = blo[2]; bu[3] = blo[3];
        bu[4] = bhi[0]; bu[5] = bhi[1]; bu[6] = bhi[2]; bu[7] = bhi[3];
        v16bf b = __builtin_bit_cast(v16bf, bu);
        acc0 = __builtin_amdgcn_wmma_f32_16x16x32_bf16(
            false, a0, false, b, (short)0, acc0, false, false);
        acc1 = __builtin_amdgcn_wmma_f32_16x16x32_bf16(
            false, a1, false, b, (short)0, acc1, false, false);
    }

    #pragma unroll
    for (int half = 0; half < 2; ++half) {
        const v8f acc = half ? acc1 : acc0;
        #pragma unroll
        for (int r = 0; r < 8; ++r) {
            const int m = m0 + half * 16 + hi * 8 + r;
            const int n = n0 + lr;
            float v = acc[r];
            if (EPI == EPI_SOFTPLUS_BIAS) {
                v += bias[n];
                v = (v > 20.0f) ? v : log1pf(__expf(v));
                ((float*)C)[(size_t)m * ldc + n] = v;
            } else if (EPI == EPI_BF16) {
                ((unsigned short*)C)[(size_t)m * ldc + n] = f32_to_bf16_rne(v);
            } else {
                ((float*)C)[(size_t)m * ldc + n] = v;
            }
        }
    }
}

// ---------------------------------------------------------------------
// Weight prep: W[K,N] f32 -> Wt[N,K] bf16, LDS-tiled 32x32 transpose.
// Uses gfx1250 async global->LDS copies when the toolchain exposes them.
// grid = (N/32, K/32), block = (32, 8).
// ---------------------------------------------------------------------
#if __has_builtin(__builtin_amdgcn_global_load_async_to_lds_b32) && \
    __has_builtin(__builtin_amdgcn_s_wait_asynccnt)
#define USE_ASYNC_LDS 1
typedef __attribute__((address_space(1))) int gas_int;
typedef __attribute__((address_space(3))) int las_int;
#else
#define USE_ASYNC_LDS 0
#endif

__global__ __launch_bounds__(256) void transpose_bf16_kernel(
    const float* __restrict__ W, unsigned short* __restrict__ Wt,
    int K, int N)
{
    __shared__ float tile[32][33];
    const int k0 = blockIdx.y * 32;
    const int n0 = blockIdx.x * 32;
    const int tx = threadIdx.x, ty = threadIdx.y;
#if USE_ASYNC_LDS
    #pragma unroll
    for (int i = 0; i < 4; ++i) {
        __builtin_amdgcn_global_load_async_to_lds_b32(
            (gas_int*)(W + (size_t)(k0 + ty + i * 8) * N + (n0 + tx)),
            (las_int*)&tile[ty + i * 8][tx],
            0, 0);
    }
    __builtin_amdgcn_s_wait_asynccnt(0);
#else
    #pragma unroll
    for (int i = 0; i < 4; ++i)
        tile[ty + i * 8][tx] = W[(size_t)(k0 + ty + i * 8) * N + (n0 + tx)];
#endif
    __syncthreads();
    #pragma unroll
    for (int i = 0; i < 4; ++i)
        Wt[(size_t)(n0 + ty + i * 8) * K + (k0 + tx)] =
            f32_to_bf16_rne(tile[tx][ty + i * 8]);
}

// f32 -> bf16 elementwise (4 per thread)
__global__ __launch_bounds__(256) void cvt_bf16_kernel(
    const float* __restrict__ src, unsigned short* __restrict__ dst)
{
    const size_t i = ((size_t)blockIdx.x * 256 + threadIdx.x) * 4;
    #pragma unroll
    for (int e = 0; e < 4; ++e) dst[i + e] = f32_to_bf16_rne(src[i + e]);
}

// ---------------------------------------------------------------------
// Causal depthwise conv1d (k=4) + bias + SiLU, reading bf16 xz.
// ---------------------------------------------------------------------
__global__ __launch_bounds__(256) void conv_silu_kernel(
    const unsigned short* __restrict__ xz, const float* __restrict__ cw,
    const float* __restrict__ cb, float* __restrict__ xs,
    unsigned short* __restrict__ xsb)
{
    const int idx = blockIdx.x * 256 + threadIdx.x;   // over BL*DINNER
    const int d = idx & (DINNER - 1);
    const int t = idx >> 12;
    const int l = t & (SEQLEN - 1);
    const int b = t >> 11;

    float acc = cb[d];
    #pragma unroll
    for (int k = 0; k < DCONV; ++k) {
        const int ls = l - (DCONV - 1) + k;
        const float xv = (ls >= 0)
            ? bf16_to_f32(xz[(size_t)(b * SEQLEN + ls) * NXZ + d]) : 0.0f;
        acc = fmaf(xv, cw[d * DCONV + k], acc);
    }
    const float s = acc / (1.0f + __expf(-acc));      // silu
    xs[idx]  = s;
    xsb[idx] = f32_to_bf16_rne(s);
}

// ---------------------------------------------------------------------
// Chunked selective scan: h_l = dA_l*h_{l-1} + (dt*u)_l*B_l (affine per state)
// Phase 1: per-chunk decay product P and local state S (zero-init h).
// ---------------------------------------------------------------------
__global__ __launch_bounds__(256) void scan_phase1(
    const float* __restrict__ dt, const float* __restrict__ proj,
    const float* __restrict__ xs, const float* __restrict__ A_log,
    float* __restrict__ P, float* __restrict__ S)
{
    const int d = blockIdx.x * 256 + threadIdx.x;
    const int c = blockIdx.y;
    const int b = blockIdx.z;

    float Ac[DSTATE], h[DSTATE], p[DSTATE];
    #pragma unroll
    for (int s = 0; s < DSTATE; ++s) {
        Ac[s] = -__expf(A_log[d * DSTATE + s]);
        h[s]  = 0.0f;
        p[s]  = 1.0f;
    }

    __shared__ float sB[DSTATE];
    for (int l = c * CHUNK; l < (c + 1) * CHUNK; ++l) {
        const size_t t = (size_t)(b * SEQLEN + l);
        if (threadIdx.x < DSTATE)
            sB[threadIdx.x] = proj[t * NPROJ + DTRANK + threadIdx.x];
        __syncthreads();
        const float dtv = dt[t * DINNER + d];
        const float du  = dtv * xs[t * DINNER + d];
        #pragma unroll
        for (int s = 0; s < DSTATE; ++s) {
            const float dA = __expf(dtv * Ac[s]);
            h[s] = fmaf(h[s], dA, du * sB[s]);
            p[s] *= dA;
        }
        __syncthreads();
    }
    const size_t o = (((size_t)b * NCHUNK + c) * DINNER + d) * DSTATE;
    #pragma unroll
    for (int s = 0; s < DSTATE; ++s) { P[o + s] = p[s]; S[o + s] = h[s]; }
}

// Phase 2: prefix over the 16 chunks -> incoming state H0 per chunk.
__global__ __launch_bounds__(256) void scan_phase2(
    const float* __restrict__ P, const float* __restrict__ S,
    float* __restrict__ H0)
{
    const int idx = blockIdx.x * 256 + threadIdx.x;  // over BATCH*DINNER*DSTATE
    const int b = idx >> 16;                         // DINNER*DSTATE = 65536
    const int r = idx & 0xFFFF;                      // d*DSTATE + s
    float h = 0.0f;
    for (int c = 0; c < NCHUNK; ++c) {
        const size_t o = ((size_t)b * NCHUNK + c) * (DINNER * DSTATE) + r;
        H0[o] = h;
        h = fmaf(P[o], h, S[o]);
    }
}

// Phase 3: replay each chunk with correct H0; fused skip + gate -> bf16.
__global__ __launch_bounds__(256) void scan_phase3(
    const float* __restrict__ dt, const float* __restrict__ proj,
    const float* __restrict__ xs, const float* __restrict__ A_log,
    const float* __restrict__ H0, const float* __restrict__ Dp,
    const unsigned short* __restrict__ xz, unsigned short* __restrict__ yg)
{
    const int d = blockIdx.x * 256 + threadIdx.x;
    const int c = blockIdx.y;
    const int b = blockIdx.z;

    float Ac[DSTATE], h[DSTATE];
    const size_t o = (((size_t)b * NCHUNK + c) * DINNER + d) * DSTATE;
    #pragma unroll
    for (int s = 0; s < DSTATE; ++s) {
        Ac[s] = -__expf(A_log[d * DSTATE + s]);
        h[s]  = H0[o + s];
    }
    const float Dd = Dp[d];

    __shared__ float sBC[2 * DSTATE];
    for (int l = c * CHUNK; l < (c + 1) * CHUNK; ++l) {
        const size_t t = (size_t)(b * SEQLEN + l);
        if (threadIdx.x < 2 * DSTATE)
            sBC[threadIdx.x] = proj[t * NPROJ + DTRANK + threadIdx.x];
        __syncthreads();
        const float dtv = dt[t * DINNER + d];
        const float u   = xs[t * DINNER + d];
        const float du  = dtv * u;
        float yv = 0.0f;
        #pragma unroll
        for (int s = 0; s < DSTATE; ++s) {
            const float dA = __expf(dtv * Ac[s]);
            h[s] = fmaf(h[s], dA, du * sBC[s]);
            yv   = fmaf(h[s], sBC[DSTATE + s], yv);
        }
        yv = fmaf(u, Dd, yv);                         // + xs*D
        const float z = bf16_to_f32(xz[t * NXZ + DINNER + d]);
        yg[t * DINNER + d] =
            f32_to_bf16_rne(yv * (z / (1.0f + __expf(-z))));
        __syncthreads();
    }
}

// ---------------------------------------------------------------------
extern "C" void kernel_launch(void* const* d_in, const int* in_sizes, int n_in,
                              void* d_out, int out_size, void* d_ws, size_t ws_size,
                              hipStream_t stream)
{
    const float* x         = (const float*)d_in[0];
    const float* in_proj_w = (const float*)d_in[1];
    const float* conv_w    = (const float*)d_in[2];
    const float* conv_b    = (const float*)d_in[3];
    const float* x_proj_w  = (const float*)d_in[4];
    const float* dt_proj_w = (const float*)d_in[5];
    const float* dt_proj_b = (const float*)d_in[6];
    const float* A_log     = (const float*)d_in[7];
    const float* Dp        = (const float*)d_in[8];
    const float* out_proj_w= (const float*)d_in[9];
    float* out = (float*)d_out;

    // ---- workspace carve-out ----
    char* ws = (char*)d_ws;
    size_t off = 0;
    auto carve = [&](size_t bytes) -> void* {
        void* p = (void*)(ws + off);
        off = (off + bytes + 255) & ~(size_t)255;
        return p;
    };
    unsigned short* xzb  = (unsigned short*)carve((size_t)BL * NXZ    * 2); // 64 MiB
    float*          xs   = (float*)         carve((size_t)BL * DINNER * 4); // 64 MiB
    unsigned short* xsb  = (unsigned short*)carve((size_t)BL * DINNER * 2); // 32 MiB
    float*          proj = (float*)         carve((size_t)BL * NPROJ  * 4);
    float*          dt   = (float*)         carve((size_t)BL * DINNER * 4); // 64 MiB
    unsigned short* ygb  = (unsigned short*)carve((size_t)BL * DINNER * 2); // 32 MiB
    unsigned short* xb   = (unsigned short*)carve((size_t)BL * DMODEL * 2);
    unsigned short* w1t  = (unsigned short*)carve((size_t)NXZ    * DMODEL * 2);
    unsigned short* wxt  = (unsigned short*)carve((size_t)NPROJ  * DINNER * 2);
    unsigned short* wdt  = (unsigned short*)carve((size_t)DINNER * DTRANK * 2);
    unsigned short* wot  = (unsigned short*)carve((size_t)DMODEL * DINNER * 2);
    const size_t nsum = (size_t)BATCH * NCHUNK * DINNER * DSTATE;           // 8 MiB ea
    float* Pbuf = (float*)carve(nsum * 4);
    float* Sbuf = (float*)carve(nsum * 4);
    float* H0   = (float*)carve(nsum * 4);

    const dim3 blk(256);
    const dim3 tblk(32, 8);

    // ---- 0) weight prep ----
    transpose_bf16_kernel<<<dim3(NXZ / 32,    DMODEL / 32), tblk, 0, stream>>>(in_proj_w,  w1t, DMODEL, NXZ);
    transpose_bf16_kernel<<<dim3(NPROJ / 32,  DINNER / 32), tblk, 0, stream>>>(x_proj_w,   wxt, DINNER, NPROJ);
    transpose_bf16_kernel<<<dim3(DINNER / 32, DTRANK / 32), tblk, 0, stream>>>(dt_proj_w,  wdt, DTRANK, DINNER);
    transpose_bf16_kernel<<<dim3(DMODEL / 32, DINNER / 32), tblk, 0, stream>>>(out_proj_w, wot, DINNER, DMODEL);
    cvt_bf16_kernel<<<dim3((size_t)BL * DMODEL / 1024), blk, 0, stream>>>(x, xb);

    // ---- 1) in_proj -> xz (bf16) ----
    wmma_gemm<EPI_BF16, true><<<dim3(NXZ / 32, BL / 128), blk, 0, stream>>>(
        xb, DMODEL, w1t, DMODEL, xzb, NXZ, nullptr, DMODEL);

    // ---- 2) conv + SiLU -> xs (f32 + bf16) ----
    conv_silu_kernel<<<dim3((size_t)BL * DINNER / 256), blk, 0, stream>>>(
        xzb, conv_w, conv_b, xs, xsb);

    // ---- 3) x_proj -> proj (f32) ----
    wmma_gemm<EPI_F32, true><<<dim3(NPROJ / 32, BL / 128), blk, 0, stream>>>(
        xsb, DINNER, wxt, DINNER, proj, NPROJ, nullptr, DINNER);

    // ---- 4) dt = softplus(proj[:, :64] @ Wdt + b) ----
    wmma_gemm<EPI_SOFTPLUS_BIAS, false><<<dim3(DINNER / 32, BL / 128), blk, 0, stream>>>(
        proj, NPROJ, wdt, DTRANK, dt, DINNER, dt_proj_b, DTRANK);

    // ---- 5) chunked selective scan + fused gate -> yg (bf16) ----
    scan_phase1<<<dim3(DINNER / 256, NCHUNK, BATCH), blk, 0, stream>>>(
        dt, proj, xs, A_log, Pbuf, Sbuf);
    scan_phase2<<<dim3((unsigned)(nsum / 256)), blk, 0, stream>>>(Pbuf, Sbuf, H0);
    scan_phase3<<<dim3(DINNER / 256, NCHUNK, BATCH), blk, 0, stream>>>(
        dt, proj, xs, A_log, H0, Dp, xzb, ygb);

    // ---- 6) out_proj -> out (f32) ----
    wmma_gemm<EPI_F32, true><<<dim3(DMODEL / 32, BL / 128), blk, 0, stream>>>(
        ygb, DINNER, wot, DINNER, out, DMODEL, nullptr, DINNER);
}
